// DetNet_12317966205385
// MI455X (gfx1250) — 8-bit matrix operand order repaired and checked by reading, compiled, not run on hardware
//
#include <hip/hip_runtime.h>
#include <hip/hip_bf16.h>

#define WDET 8
#define NDET 1024
#define MDET (WDET * NDET)
#define NMS_T 0.3f
#define MASK_WORDS (MDET / 32)
#define NTILES (MDET / 16)

typedef __attribute__((ext_vector_type(8))) int v8i;

#if defined(__has_builtin)
#if __has_builtin(__builtin_amdgcn_global_load_async_to_lds_b128) && \
    __has_builtin(__builtin_amdgcn_global_load_async_to_lds_b32)  && \
    __has_builtin(__builtin_amdgcn_s_wait_asynccnt)
#define USE_ASYNC_LDS 1
#endif
#endif

#if defined(USE_ASYNC_LDS)
typedef int async_v4i __attribute__((vector_size(16)));
typedef __attribute__((address_space(1))) async_v4i* g_v4i_p;
typedef __attribute__((address_space(3))) async_v4i* l_v4i_p;
typedef __attribute__((address_space(1))) int*       g_i32_p;
typedef __attribute__((address_space(3))) int*       l_i32_p;
#endif

__global__ void detnms_transform(const float* __restrict__ det,
                                 const float* __restrict__ off,
                                 const float* __restrict__ scl,
                                 const float* __restrict__ bnd,
                                 float* __restrict__ d_t,
                                 float* __restrict__ scores) {
    int m = blockIdx.x * blockDim.x + threadIdx.x;
    if (m >= MDET) return;
    int w = m >> 10;
    const float* e = det + (size_t)m * 5;
    float e0 = e[0], e1 = e[1], e2 = e[2], e3 = e[3], e4 = e[4];
    float left   = bnd[w * 4 + 0];
    float right  = bnd[w * 4 + 1];
    float top    = bnd[w * 4 + 2];
    float bottom = bnd[w * 4 + 3];
    bool valid = (e1 < right) && (e1 > left) && (e2 < bottom) && (e2 > top);
    float d0 = off[w * 5 + 0] + e0 * scl[w * 5 + 0];
    float d1 = off[w * 5 + 1] + e1 * scl[w * 5 + 1];
    float d2 = off[w * 5 + 2] + e2 * scl[w * 5 + 2];
    float d3 = off[w * 5 + 3] + e3 * scl[w * 5 + 3];
    float d4 = off[w * 5 + 4] + e4 * scl[w * 5 + 4];
    float* o = d_t + (size_t)m * 5;
    o[0] = d0; o[1] = d1; o[2] = d2; o[3] = d3; o[4] = d4;
    scores[m] = valid ? d0 : -1.0f;
}

__global__ __launch_bounds__(1024) void detnms_sort(const float* __restrict__ scores,
                                                    int* __restrict__ order) {
    extern __shared__ __align__(16) char smem[];
    float* key = (float*)smem;
    int*   idx = (int*)(smem + MDET * sizeof(float));

    for (int i = threadIdx.x; i < MDET; i += 1024) {
        key[i] = -scores[i];
        idx[i] = i;
    }
    __syncthreads();

    for (int k = 2; k <= MDET; k <<= 1) {
        for (int j = k >> 1; j > 0; j >>= 1) {
            for (int i = threadIdx.x; i < MDET; i += 1024) {
                int l = i ^ j;
                if (l > i) {
                    bool up = ((i & k) == 0);
                    float ki = key[i], kl = key[l];
                    if ((ki > kl) == up) {
                        key[i] = kl; key[l] = ki;
                        int t = idx[i]; idx[i] = idx[l]; idx[l] = t;
                    }
                }
            }
            __syncthreads();
        }
    }
    for (int i = threadIdx.x; i < MDET; i += 1024) order[i] = idx[i];
}

__global__ void detnms_gather(const float* __restrict__ d_t,
                              const float* __restrict__ scores,
                              const int* __restrict__ order,
                              float* __restrict__ d_s,
                              float* __restrict__ sscore,
                              float4* __restrict__ boxes,
                              float* __restrict__ area) {
    int p = blockIdx.x * blockDim.x + threadIdx.x;
    if (p >= MDET) return;
    int src = order[p];
    const float* s = d_t + (size_t)src * 5;
    float d0 = s[0], d1 = s[1], d2 = s[2], d3 = s[3], d4 = s[4];
    float* o = d_s + (size_t)p * 5;
    o[0] = d0; o[1] = d1; o[2] = d2; o[3] = d3; o[4] = d4;
    sscore[p] = scores[src];
    float x1 = d1 - d3 * 0.5f, y1 = d2 - d4 * 0.5f;
    float x2 = d1 + d3 * 0.5f, y2 = d2 + d4 * 0.5f;
    boxes[p] = make_float4(x1, y1, x2, y2);
    area[p]  = fmaxf(x2 - x1, 0.0f) * fmaxf(y2 - y1, 0.0f);
}

__global__ __launch_bounds__(256) void detnms_mask(const float4* __restrict__ boxes,
                                                   const float* __restrict__ area,
                                                   unsigned* __restrict__ maskT) {
    extern __shared__ __align__(16) char smem[];
    float4* lbox  = (float4*)smem;
    float*  larea = (float*)(smem + MDET * sizeof(float4));

#if defined(USE_ASYNC_LDS)
    for (int i = threadIdx.x; i < MDET; i += 256) {
        __builtin_amdgcn_global_load_async_to_lds_b128(
            (g_v4i_p)(boxes + i), (l_v4i_p)(lbox + i), 0, 0);
        __builtin_amdgcn_global_load_async_to_lds_b32(
            (g_i32_p)(area + i), (l_i32_p)(larea + i), 0, 0);
    }
    __builtin_amdgcn_s_wait_asynccnt(0);
    __syncthreads();
#else
    for (int i = threadIdx.x; i < MDET; i += 256) {
        lbox[i]  = boxes[i];
        larea[i] = area[i];
    }
    __syncthreads();
#endif

    int j = blockIdx.x;
    float4 bj = lbox[j];
    float  aj = larea[j];

    unsigned bits = 0u;
    int base = threadIdx.x * 32;
    #pragma unroll 4
    for (int b = 0; b < 32; ++b) {
        int i = base + b;
        if (i > j) {
            float4 bi = lbox[i];
            float iw = fminf(bj.z, bi.z) - fmaxf(bj.x, bi.x);
            float ih = fminf(bj.w, bi.w) - fmaxf(bj.y, bi.y);
            iw = fmaxf(iw, 0.0f); ih = fmaxf(ih, 0.0f);
            float inter = iw * ih;
            float uni   = aj + larea[i] - inter;
            float iou   = inter / fmaxf(uni, 1e-9f);
            if (iou > NMS_T) bits |= (1u << b);
        }
    }
    maskT[(size_t)j * MASK_WORDS + threadIdx.x] = bits;
}

__device__ __forceinline__ unsigned spread4(unsigned h, int s) {
    return ((h >> s) & 1u)
         | (((h >> (s + 1)) & 1u) << 8)
         | (((h >> (s + 2)) & 1u) << 16)
         | (((h >> (s + 3)) & 1u) << 24);
}

__global__ __launch_bounds__(32) void detnms_scan(const float4* __restrict__ boxes,
                                                  const float* __restrict__ area,
                                                  const float* __restrict__ sscore,
                                                  const unsigned* __restrict__ maskT,
                                                  int* __restrict__ keepOut) {
    __shared__ __align__(4) unsigned char keepL[MDET];
    const unsigned* keepW = (const unsigned*)keepL;

    int lane = threadIdx.x;
    int col  = lane & 15;

    for (int i = lane; i < MDET; i += 32) keepL[i] = 0;
    __syncthreads();

    for (int c = 0; c < NTILES; ++c) {
        int cbase = c * 16;
        v8i acc = {};

        int ngroups = (c + 3) >> 2;
        int klo = (lane < 16) ? 0 : 8;
        for (int g = 0; g < ngroups; ++g) {
            int gb = g * 64;

            v8i A;
            int wbase = (gb >> 2) + (klo >> 2);
            A[0] = (int)keepW[wbase + 0];
            A[1] = (int)keepW[wbase + 1];
            A[2] = (int)keepW[wbase + 4];
            A[3] = (int)keepW[wbase + 5];
            A[4] = (int)keepW[wbase + 8];
            A[5] = (int)keepW[wbase + 9];
            A[6] = (int)keepW[wbase + 12];
            A[7] = (int)keepW[wbase + 13];

            v8i B;
            for (int r = 0; r < 8; ++r) {
                unsigned v = 0u;
                for (int q = 0; q < 4; ++q) {
                    int kk = gb + klo + (r >> 1) * 16 + (r & 1) * 4 + q;
                    unsigned wv = maskT[(size_t)kk * MASK_WORDS + (c >> 1)];
                    v |= ((wv >> (((c & 1) << 4) + col)) & 1u) << (8 * q);
                }
                B[r] = (int)v;
            }

            acc = __builtin_amdgcn_wmma_i32_16x16x64_iu8(
                       false, A,  false, B, acc,
                       false,  false);
        }
        int supAcc = acc[0];

        int i = cbase + col;
        float4 bi = boxes[i];
        float  ai = area[i];
        float  si = sscore[i];
        int myKeep = 0;
        int supIn  = 0;
        for (int t = 0; t < 16; ++t) {
            int dec = 0;
            if (lane == t)
                dec = (supAcc == 0 && supIn == 0 && si > 0.0f) ? 1 : 0;
            int kt = __shfl(dec, t);
            if (lane == t) myKeep = dec;
            if (kt) {
                float x1t = __shfl(bi.x, t);
                float y1t = __shfl(bi.y, t);
                float x2t = __shfl(bi.z, t);
                float y2t = __shfl(bi.w, t);
                float at  = __shfl(ai,  t);
                if (lane < 16 && col > t) {
                    float iw = fminf(x2t, bi.z) - fmaxf(x1t, bi.x);
                    float ih = fminf(y2t, bi.w) - fmaxf(y1t, bi.y);
                    iw = fmaxf(iw, 0.0f); ih = fmaxf(ih, 0.0f);
                    float inter = iw * ih;
                    float uni   = at + ai - inter;
                    if (inter / fmaxf(uni, 1e-9f) > NMS_T) supIn += 1;
                }
            }
        }
        if (lane < 16) {
            keepL[cbase + lane] = (unsigned char)myKeep;
            keepOut[cbase + lane] = myKeep;
        }
        __syncthreads();
    }
}

__global__ void detnms_finalize(const float* __restrict__ d_s,
                                const int* __restrict__ keep,
                                float* __restrict__ out) {
    int p = blockIdx.x * blockDim.x + threadIdx.x;
    if (p >= MDET) return;
    float kf = keep[p] ? 1.0f : 0.0f;
    const float* s = d_s + (size_t)p * 5;
    float* o = out + (size_t)p * 5;
    o[0] = s[0] * kf; o[1] = s[1] * kf; o[2] = s[2] * kf;
    o[3] = s[3] * kf; o[4] = s[4] * kf;
}

#define OFF_DT    ((size_t)0)
#define OFF_SC    (OFF_DT   + (size_t)MDET * 5 * 4)
#define OFF_ORD   (OFF_SC   + (size_t)MDET * 4)
#define OFF_DS    (OFF_ORD  + (size_t)MDET * 4)
#define OFF_SS    (OFF_DS   + (size_t)MDET * 5 * 4)
#define OFF_BOX   (OFF_SS   + (size_t)MDET * 4)
#define OFF_AREA  (OFF_BOX  + (size_t)MDET * 16)
#define OFF_MASK  (OFF_AREA + (size_t)MDET * 4)
#define OFF_KEEP  (OFF_MASK + (size_t)MDET * MASK_WORDS * 4)

extern "C" void kernel_launch(void* const* d_in, const int* in_sizes, int n_in,
                              void* d_out, int out_size, void* d_ws, size_t ws_size,
                              hipStream_t stream) {
    const float* det = (const float*)d_in[0];
    const float* off = (const float*)d_in[1];
    const float* scl = (const float*)d_in[2];
    const float* bnd = (const float*)d_in[3];

    char* ws = (char*)d_ws;
    float*    d_t    = (float*)(ws + OFF_DT);
    float*    scores = (float*)(ws + OFF_SC);
    int*      order  = (int*)(ws + OFF_ORD);
    float*    d_s    = (float*)(ws + OFF_DS);
    float*    sscore = (float*)(ws + OFF_SS);
    float4*   boxes  = (float4*)(ws + OFF_BOX);
    float*    area   = (float*)(ws + OFF_AREA);
    unsigned* maskT  = (unsigned*)(ws + OFF_MASK);
    int*      keep   = (int*)(ws + OFF_KEEP);
    float*    out    = (float*)d_out;

    detnms_transform<<<MDET / 256, 256, 0, stream>>>(det, off, scl, bnd, d_t, scores);

    detnms_sort<<<1, 1024, MDET * 8, stream>>>(scores, order);

    detnms_gather<<<MDET / 256, 256, 0, stream>>>(d_t, scores, order,
                                                  d_s, sscore, boxes, area);

    detnms_mask<<<MDET, 256, MDET * 20, stream>>>(boxes, area, maskT);

    detnms_scan<<<1, 32, 0, stream>>>(boxes, area, sscore, maskT, keep);

    detnms_finalize<<<MDET / 256, 256, 0, stream>>>(d_s, keep, out);
}
